// DotProductAttention_32341103739228
// MI455X (gfx1250) — matrix operand order repaired and checked by reading, not compiled, not run on hardware
//
#include <hip/hip_runtime.h>

// Flash-attention: N=8, S=2048, D=V=64, fp32 I/O, bf16 WMMA math (CDNA5).
// Mask: key < prefix_len[n]  OR  key == query (diagonal).
//
// - Transposed-score form: C = S^T = K_tile x Q^T; softmax in-lane + 1 xor-16.
// - exp(S^T) C-fragment packs directly into P^T B-fragment (key-row permute).
// - V stored transposed in LDS -> V^T A-fragments are b128 LDS loads.
// - Double-buffered LDS, software-pipelined global loads, 1 barrier/tile.
// - Uniform masked/unmasked tile split; scale folded into Q (exact, pow2).

#define N_HEADS  8
#define SEQ      2048
#define DIM      64
#define KT       32            // keys per tile (== WMMA K for bf16)
#define QW       16            // queries per wave
#define WAVES    4
#define QB       (QW*WAVES)    // 64 queries per block
#define KSTRIDE  (DIM + 8)     // Kl row stride: 72 elems = 144 B (16B aligned)
#define VTSTRIDE (KT + 8)      // Vt row stride: 40 elems =  80 B (16B aligned)

typedef __attribute__((ext_vector_type(16))) __bf16 v16bf;
typedef __attribute__((ext_vector_type(8)))  __bf16 v8bf;
typedef __attribute__((ext_vector_type(4)))  __bf16 v4bf;
typedef __attribute__((ext_vector_type(8)))  float  v8f;

#define SHUF16(LO, HI) __builtin_shufflevector((LO), (HI), 0,1,2,3,4,5,6,7,8,9,10,11,12,13,14,15)

__device__ __forceinline__ v16bf frag_b128(const __bf16* p) {
    // elements [0..7] at p, [8..15] at p+16 (A-layout 8-interleave), both 16B aligned
    return SHUF16(*(const v8bf*)p, *(const v8bf*)(p + 16));
}

// One 32-key tile: S^T WMMAs, online softmax, O^T += V^T x P^T.
template<bool MASKED>
__device__ __forceinline__ void attn_tile(
    const __bf16* __restrict__ Klb,   // [KT][KSTRIDE] keys-major
    const __bf16* __restrict__ Vtb,   // [DIM][VTSTRIDE] v-major (transposed)
    int s0, int myq, int prefix, int h, int cn, int keyrow0,
    const v16bf& qf0, const v16bf& qf1,
    float& rmax, float& rsum, v8f (&oacc)[4])
{
    // ---- S^T = K_tile x Q^T ---- (A rows = permuted keys, d contiguous)
    const __bf16* r0 = &Klb[ keyrow0      * KSTRIDE + 8 * h];
    const __bf16* r1 = &Klb[(keyrow0 + 8) * KSTRIDE + 8 * h];
    const v16bf ka00 = frag_b128(r0);
    const v16bf ka01 = frag_b128(r0 + 32);
    const v16bf ka10 = frag_b128(r1);
    const v16bf ka11 = frag_b128(r1 + 32);

    v8f c0, c1;
    #pragma unroll
    for (int e = 0; e < 8; ++e) { c0[e] = 0.0f; c1[e] = 0.0f; }
    c0 = __builtin_amdgcn_wmma_f32_16x16x32_bf16(false, ka00, false, qf0, (short)0, c0, false, false);
    c0 = __builtin_amdgcn_wmma_f32_16x16x32_bf16(false, ka01, false, qf1, (short)0, c0, false, false);
    c1 = __builtin_amdgcn_wmma_f32_16x16x32_bf16(false, ka10, false, qf0, (short)0, c1, false, false);
    c1 = __builtin_amdgcn_wmma_f32_16x16x32_bf16(false, ka11, false, qf1, (short)0, c1, false, false);

    // lane (h,cn): c0[r] <-> key s0+16h+r, c1[r] <-> key s0+16h+8+r, query myq
    float sv[16];
    #pragma unroll
    for (int r = 0; r < 8; ++r) {
        float a = c0[r], b = c1[r];
        if (MASKED) {
            const int key0 = s0 + 16 * h + r;
            const int key1 = key0 + 8;
            if (!((key0 < prefix) || (key0 == myq))) a = -1.0e9f;
            if (!((key1 < prefix) || (key1 == myq))) b = -1.0e9f;
        }
        sv[r] = a; sv[8 + r] = b;
    }

    // ---- online softmax: in-lane reduce + one xor-16 combine ----
    float m16 = sv[0];
    #pragma unroll
    for (int i = 1; i < 16; ++i) m16 = fmaxf(m16, sv[i]);
    m16 = fmaxf(m16, __shfl_xor(m16, 16, 32));
    const float mnew = fmaxf(rmax, m16);
    const float scl  = __expf(rmax - mnew);

    float lsum = 0.0f;
    v16bf pb;                           // P^T B-fragment: pb[e] <-> key s0+16h+e
    #pragma unroll
    for (int i = 0; i < 16; ++i) {
        const float p = __expf(sv[i] - mnew);
        lsum += p;
        pb[i] = (__bf16)p;
    }
    lsum += __shfl_xor(lsum, 16, 32);
    rsum = rsum * scl + lsum;
    rmax = mnew;
    #pragma unroll
    for (int j = 0; j < 4; ++j)
        #pragma unroll
        for (int e = 0; e < 8; ++e) oacc[j][e] *= scl;

    // ---- O^T += V^T x P^T : A = V^T fragments, contiguous key runs ----
    #pragma unroll
    for (int j = 0; j < 4; ++j) {
        const __bf16* vp = &Vtb[(16 * j + cn) * VTSTRIDE + 16 * h];
        const v16bf va = SHUF16(*(const v8bf*)vp, *(const v8bf*)(vp + 8));
        oacc[j] = __builtin_amdgcn_wmma_f32_16x16x32_bf16(false, va, false, pb, (short)0, oacc[j], false, false);
    }
}

__global__ __launch_bounds__(128, 2)
void fa_fwd_kernel(const float* __restrict__ Q,
                   const float* __restrict__ K,
                   const float* __restrict__ V,
                   const int*   __restrict__ plen,
                   float*       __restrict__ O)
{
    __shared__ __bf16 Kl[2][KT  * KSTRIDE];   // double-buffered K tile (keys-major)
    __shared__ __bf16 Vt[2][DIM * VTSTRIDE];  // double-buffered V tile (transposed)

    const int bx   = blockIdx.x;
    const int n    = bx / (SEQ / QB);
    const int qb   = (bx % (SEQ / QB)) * QB;
    const int tid  = threadIdx.x;
    const int wave = tid >> 5;
    const int lane = tid & 31;
    const int h    = lane >> 4;
    const int cn   = lane & 15;
    const int myq  = qb + wave * QW + cn;
    const int prefix = plen[n];

    const float* Qb = Q + (size_t)n * SEQ * DIM;
    const float* Kb = K + (size_t)n * SEQ * DIM;
    const float* Vb = V + (size_t)n * SEQ * DIM;

    // ---- Q^T B-fragments, scale 1/sqrt(64)=0.125 folded in (exact pow2) ----
    v16bf qf0, qf1;
    {
        const float* qrow = Qb + (size_t)myq * DIM;
        #pragma unroll
        for (int v = 0; v < 8; ++v) {
            const int d = 2 * v + 8 * h + ((v >> 2) << 3); // B-layout K offset (table order, matches A)
            float2 x0 = *(const float2*)(qrow + d);
            float2 x1 = *(const float2*)(qrow + 32 + d);
            qf0[2*v]   = (__bf16)(x0.x * 0.125f);  qf0[2*v+1] = (__bf16)(x0.y * 0.125f);
            qf1[2*v]   = (__bf16)(x1.x * 0.125f);  qf1[2*v+1] = (__bf16)(x1.y * 0.125f);
        }
    }

    float rmax = -3.0e38f, rsum = 0.0f;
    v8f oacc[4];
    #pragma unroll
    for (int j = 0; j < 4; ++j)
        #pragma unroll
        for (int e = 0; e < 8; ++e) oacc[j][e] = 0.0f;

    // key->A-row permutation (subtile st row m holds key 16*(m>>3)+8*st+(m&7))
    const int keyrow0 = 16 * (cn >> 3) + (cn & 7);

    // ---- explicit needed-tile list: prefix tiles + <=2 diagonal tiles ----
    const int prefixEnd = (prefix + KT - 1) & ~(KT - 1);
    const int nPref     = prefixEnd >> 5;                       // /KT
    const int nEx       = (qb >= prefixEnd) ? 2 : ((qb + KT >= prefixEnd) ? 1 : 0);
    const int nT        = nPref + nEx;

    auto tile_s0 = [&](int i) -> int {
        if (i < nPref) return i * KT;
        return (qb >= prefixEnd) ? (qb + (i - nPref) * KT) : (qb + KT);
    };

    // ---- pipelined tile loop: global loads of tile i+1 overlap WMMAs of i ----
    const int lrow = tid >> 2;             // 0..31 (key row)
    const int lc4  = (tid & 3) * 16;       // 0,16,32,48 (column base)
    float4 kreg[4], vreg[4];

    auto load_regs = [&](int s0t) {
        const float4* ks = (const float4*)(Kb + (size_t)(s0t + lrow) * DIM + lc4);
        const float4* vs = (const float4*)(Vb + (size_t)(s0t + lrow) * DIM + lc4);
        #pragma unroll
        for (int i = 0; i < 4; ++i) { kreg[i] = ks[i]; vreg[i] = vs[i]; }
    };
    auto store_lds = [&](int buf) {
        #pragma unroll
        for (int i = 0; i < 4; ++i) {
            v4bf kk;
            kk[0] = (__bf16)kreg[i].x; kk[1] = (__bf16)kreg[i].y;
            kk[2] = (__bf16)kreg[i].z; kk[3] = (__bf16)kreg[i].w;
            *(v4bf*)&Kl[buf][lrow * KSTRIDE + lc4 + 4 * i] = kk;
            const int c = lc4 + 4 * i;                 // transposed V store
            Vt[buf][(c + 0) * VTSTRIDE + lrow] = (__bf16)vreg[i].x;
            Vt[buf][(c + 1) * VTSTRIDE + lrow] = (__bf16)vreg[i].y;
            Vt[buf][(c + 2) * VTSTRIDE + lrow] = (__bf16)vreg[i].z;
            Vt[buf][(c + 3) * VTSTRIDE + lrow] = (__bf16)vreg[i].w;
        }
    };

    load_regs(tile_s0(0));
    store_lds(0);
    __syncthreads();

    for (int i = 0; i < nT; ++i) {
        const int  cur  = i & 1;
        const bool more = (i + 1 < nT);
        if (more) load_regs(tile_s0(i + 1));   // in flight during WMMAs below

        const int sCur = tile_s0(i);
        if (sCur + KT <= prefix)               // block-uniform branch
            attn_tile<false>(Kl[cur], Vt[cur], sCur, myq, prefix, h, cn, keyrow0,
                             qf0, qf1, rmax, rsum, oacc);
        else
            attn_tile<true >(Kl[cur], Vt[cur], sCur, myq, prefix, h, cn, keyrow0,
                             qf0, qf1, rmax, rsum, oacc);

        if (more) store_lds(1 - cur);
        __syncthreads();
    }

    // ---- epilogue: normalize, contiguous b128 stores ----
    const float inv = 1.0f / rsum;             // diagonal guarantees rsum > 0
    float* orow = O + (size_t)n * SEQ * DIM + (size_t)myq * DIM + 8 * h;
    #pragma unroll
    for (int j = 0; j < 4; ++j) {
        float4 lo, hi;
        lo.x = oacc[j][0]*inv; lo.y = oacc[j][1]*inv; lo.z = oacc[j][2]*inv; lo.w = oacc[j][3]*inv;
        hi.x = oacc[j][4]*inv; hi.y = oacc[j][5]*inv; hi.z = oacc[j][6]*inv; hi.w = oacc[j][7]*inv;
        *(float4*)(orow + 16 * j)     = lo;    // v = 16j + 8h + {0..3}
        *(float4*)(orow + 16 * j + 4) = hi;    // v = 16j + 8h + {4..7}
    }
}

extern "C" void kernel_launch(void* const* d_in, const int* in_sizes, int n_in,
                              void* d_out, int out_size, void* d_ws, size_t ws_size,
                              hipStream_t stream) {
    const float* Q    = (const float*)d_in[0];
    const float* K    = (const float*)d_in[1];
    const float* V    = (const float*)d_in[2];
    const int*   plen = (const int*)d_in[3];
    float*       O    = (float*)d_out;

    dim3 grid(N_HEADS * (SEQ / QB));
    dim3 block(128);
    fa_fwd_kernel<<<grid, block, 0, stream>>>(Q, K, V, plen, O);
}
